// Renderer_12171937317459
// MI455X (gfx1250) — matrix operand order repaired and checked by reading, not compiled, not run on hardware
//
#include <hip/hip_runtime.h>

// ---------------- problem constants ----------------
#define BATCH 8
#define NPTS  2048
#define KNN   10
#define BN    (BATCH * NPTS)
#define ROWSE (BN * KNN)          // 163840 edges

typedef __attribute__((ext_vector_type(16))) _Float16 v16h;
typedef __attribute__((ext_vector_type(8)))  _Float16 v8h;
typedef __attribute__((ext_vector_type(8)))  float    v8f;

__device__ __forceinline__ float preluf(float v, float a) {
    return v >= 0.0f ? v : a * v;
}

// =====================================================================
// Kernel 1: x = x0 + d*r, sdf = |x|-1, grad = x/|x|; build li [BN,10]
// and write x / sdf / grad directly into their d_out slots.
// =====================================================================
__global__ void init_kernel(const float* __restrict__ d,
                            const float* __restrict__ x0,
                            const float* __restrict__ r,
                            float* __restrict__ li,
                            float* __restrict__ out) {
    int t = blockIdx.x * blockDim.x + threadIdx.x;
    if (t >= BN) return;
    int b = t / NPTS;
    float dd = d[t];
    float xx[3], rr[3], gg[3];
#pragma unroll
    for (int c = 0; c < 3; ++c) {
        rr[c] = r[t * 3 + c];
        xx[c] = x0[b * 3 + c] + dd * rr[c];
    }
    float nrm = sqrtf(xx[0] * xx[0] + xx[1] * xx[1] + xx[2] * xx[2]);
    float sdf = nrm - 1.0f;
#pragma unroll
    for (int c = 0; c < 3; ++c) gg[c] = xx[c] / nrm;

    float* lr = li + (size_t)t * 10;
#pragma unroll
    for (int c = 0; c < 3; ++c) { lr[c] = xx[c]; lr[3 + c] = rr[c]; lr[7 + c] = gg[c]; }
    lr[6] = sdf;

    // d_out layout: [out0|out1|out2|out3 | x(3BN) | sdf(BN) | grad(3BN)]
#pragma unroll
    for (int c = 0; c < 3; ++c) {
        out[4 * BN + (size_t)t * 3 + c] = xx[c];
        out[8 * BN + (size_t)t * 3 + c] = gg[c];
    }
    out[7 * BN + t] = sdf;
}

// =====================================================================
// Kernel 2: exact fp32 kNN (top-KNN smallest sq-distances, incl. self).
// One thread per query row; 128-row LDS tiles of candidates.
// Tie behavior matches jax.lax.top_k (lower index first).
// =====================================================================
template <int C>
__global__ void knn_kernel(const float* __restrict__ src, int ld, int coff,
                           int* __restrict__ idxOut) {
    __shared__ float tile[128 * C];
    __shared__ float tsq[128];
    int b = blockIdx.y;
    int i = blockIdx.x * 128 + threadIdx.x;
    const float* base = src + (size_t)b * NPTS * ld + coff;

    float xi[C];
    float sqi = 0.0f;
#pragma unroll
    for (int c = 0; c < C; ++c) {
        xi[c] = base[(size_t)i * ld + c];
        sqi += xi[c] * xi[c];
    }

    float dk[KNN];
    int   ik[KNN];
#pragma unroll
    for (int m = 0; m < KNN; ++m) { dk[m] = 3.4e38f; ik[m] = 0; }

    for (int j0 = 0; j0 < NPTS; j0 += 128) {
        __syncthreads();
        int jt = j0 + threadIdx.x;
        float s = 0.0f;
#pragma unroll
        for (int c = 0; c < C; ++c) {
            float v = base[(size_t)jt * ld + c];
            tile[threadIdx.x * C + c] = v;
            s += v * v;
        }
        tsq[threadIdx.x] = s;
        __syncthreads();

        for (int jj = 0; jj < 128; ++jj) {
            float dot = 0.0f;
#pragma unroll
            for (int c = 0; c < C; ++c) dot += xi[c] * tile[jj * C + c];
            float dist = sqi + tsq[jj] - 2.0f * dot;
            if (dist < dk[KNN - 1]) {
                dk[KNN - 1] = dist;
                ik[KNN - 1] = j0 + jj;
#pragma unroll
                for (int m = KNN - 1; m > 0; --m) {      // bubble up, strict <
                    float a0 = dk[m - 1], b0 = dk[m];
                    int   a1 = ik[m - 1], b1 = ik[m];
                    bool sw = b0 < a0;
                    dk[m - 1] = sw ? b0 : a0;  ik[m - 1] = sw ? b1 : a1;
                    dk[m]     = sw ? a0 : b0;  ik[m]     = sw ? a1 : b1;
                }
            }
        }
    }
    int* op = idxOut + ((size_t)b * NPTS + i) * KNN;
#pragma unroll
    for (int m = 0; m < KNN; ++m) op[m] = ik[m];
}

// =====================================================================
// Weight convert: f32 [Nout,Kin] -> f16 [Npad,Kpad], zero padded.
// =====================================================================
__global__ void wconv_kernel(const float* __restrict__ src, _Float16* __restrict__ dst,
                             int Nout, int Kin, int Npad, int Kpad) {
    int t = blockIdx.x * blockDim.x + threadIdx.x;
    int total = Npad * Kpad;
    if (t >= total) return;
    int n = t / Kpad, k = t - n * Kpad;
    float v = (n < Nout && k < Kin) ? src[(size_t)n * Kin + k] : 0.0f;
    dst[t] = (_Float16)v;
}

// =====================================================================
// Edge-feature gather: ef16[edge, k] = k<C ? src[nbr]-src[i] : src[i] (pad 0)
// diff computed in fp32, rounded once to f16.
// =====================================================================
__global__ void edge_gather(const float* __restrict__ src, int ld, int coff, int C, int KP,
                            const int* __restrict__ nbr, _Float16* __restrict__ ef) {
    int t = blockIdx.x * blockDim.x + threadIdx.x;
    long total = (long)ROWSE * KP;
    if (t >= total) return;
    int edge = t / KP;
    int k = t - edge * KP;
    int b = edge / (NPTS * KNN);
    int rem = edge - b * (NPTS * KNN);
    int i = rem / KNN;
    int nb = nbr[edge];
    const float* ci = src + ((size_t)b * NPTS + i)  * ld + coff;
    const float* cn = src + ((size_t)b * NPTS + nb) * ld + coff;
    float v = 0.0f;
    if (k < C)          v = cn[k] - ci[k];
    else if (k < 2 * C) v = ci[k - C];
    ef[t] = (_Float16)v;
}

// =====================================================================
// WMMA GEMM: out = PReLU(A16[rows,Kin] @ W16[Npad,Kin]^T).
// All operands pre-converted f16, Kin % 32 == 0, Npad % (16*NT) == 0,
// rows % 16 == 0  ->  guard-free, software-pipelined inner loop with
// vectorized b128 fragment loads (next k-tile prefetched into registers
// before the current tile's WMMAs, so s_wait_loadcnt need not fully drain).
// Fragment layouts per CDNA5 ISA 7.12.2:
//   A 16x32 f16 : row = lane&15,  k = 16*(e>>3) + 8*(lane>>4) + (e&7)
//   B 32x16 f16 : col = lane&15,  k = 16*(e>>3) + 8*(lane>>4) + (e&7)
//   C/D 16x16   : row = v + 8*(lane>>4), col = lane&15
// Each wave computes a 16 x (16*NT) tile, reusing A across NT wmmas.
// =====================================================================
template <int NT>
__global__ void gemm16(const _Float16* __restrict__ A, int lda,
                       const _Float16* __restrict__ W, int Kin, int Npad,
                       const float* __restrict__ pa,
                       float* __restrict__ outF, int ldo, int NoutStore,
                       _Float16* __restrict__ out16, int ldo16,
                       int rows) {
    int wave = threadIdx.x >> 5;
    int lane = threadIdx.x & 31;
    int NG = Npad / (16 * NT);
    int tile = blockIdx.x * (blockDim.x >> 5) + wave;
    int total = (rows >> 4) * NG;
    if (tile >= total) return;
    int tm = tile / NG, ng = tile - tm * NG;
    int m0 = tm << 4;

    const _Float16* ar = A + (size_t)(m0 + (lane & 15)) * lda + ((lane >> 4) << 3);
    int colLane = ng * NT * 16 + (lane & 15);
    int kBoff = (lane >> 4) << 3;

    auto loadA = [&](int kb) -> v16h {
        v8h alo = *(const v8h*)(ar + kb);
        v8h ahi = *(const v8h*)(ar + kb + 16);
        return __builtin_shufflevector(alo, ahi, 0, 1, 2, 3, 4, 5, 6, 7,
                                                 8, 9, 10, 11, 12, 13, 14, 15);
    };
    auto loadB = [&](int kb, int nt) -> v16h {
        const _Float16* wr = W + (size_t)(colLane + nt * 16) * Kin + kb + kBoff;
        v8h blo = *(const v8h*)wr;
        v8h bhi = *(const v8h*)(wr + 16);
        return __builtin_shufflevector(blo, bhi, 0, 1, 2, 3, 4, 5, 6, 7,
                                                 8, 9, 10, 11, 12, 13, 14, 15);
    };

    v8f acc[NT];
#pragma unroll
    for (int nt = 0; nt < NT; ++nt) acc[nt] = {};

    // ---- software pipeline: stage 0 loads, then overlap next-tile loads ----
    v16h aCur = loadA(0);
    v16h bCur[NT];
#pragma unroll
    for (int nt = 0; nt < NT; ++nt) bCur[nt] = loadB(0, nt);

    for (int kb = 0; kb < Kin; kb += 32) {
        int kn = (kb + 32 < Kin) ? (kb + 32) : 0;   // last iter re-reads tile 0 (dead)
        v16h aNxt = loadA(kn);
        v16h bNxt[NT];
#pragma unroll
        for (int nt = 0; nt < NT; ++nt) bNxt[nt] = loadB(kn, nt);

#pragma unroll
        for (int nt = 0; nt < NT; ++nt)
            acc[nt] = __builtin_amdgcn_wmma_f32_16x16x32_f16(false, aCur, false, bCur[nt],
                                                             (short)0, acc[nt],
                                                             false, false);
        aCur = aNxt;
#pragma unroll
        for (int nt = 0; nt < NT; ++nt) bCur[nt] = bNxt[nt];
    }

    float alpha = pa[0];
    int drow = m0 + ((lane >> 4) << 3);
#pragma unroll
    for (int nt = 0; nt < NT; ++nt) {
        int col = ng * NT * 16 + nt * 16 + (lane & 15);
#pragma unroll
        for (int v = 0; v < 8; ++v) {
            float val = preluf(acc[nt][v], alpha);
            if (col < NoutStore) {
                if (outF)  outF[(size_t)(drow + v) * ldo + col] = val;
                if (out16) out16[(size_t)(drow + v) * ldo16 + col] = (_Float16)val;
            }
        }
    }
}

// =====================================================================
// Reductions, h-builder, scatter
// =====================================================================
__global__ void maxk_kernel(const float* __restrict__ eo,      // [ROWSE, 64] f32
                            float* __restrict__ feat,          // [BN, 192] f32
                            _Float16* __restrict__ feat16,     // [BN, 192] f16
                            int coff) {
    int t = blockIdx.x * blockDim.x + threadIdx.x;
    if (t >= BN * 64) return;
    int point = t >> 6;
    int c = t & 63;
    float m = -3.4e38f;
#pragma unroll
    for (int kk = 0; kk < KNN; ++kk)
        m = fmaxf(m, eo[((size_t)point * KNN + kk) * 64 + c]);
    feat[(size_t)point * 192 + coff + c] = m;
    feat16[(size_t)point * 192 + coff + c] = (_Float16)m;
}

__global__ void gmax_kernel(const float* __restrict__ x5,   // [BN, 1024]
                            float* __restrict__ gmax) {     // [B, 1024]
    int t = blockIdx.x * blockDim.x + threadIdx.x;
    if (t >= BATCH * 1024) return;
    int b = t >> 10;
    int c = t & 1023;
    float m = -3.4e38f;
    for (int i = 0; i < NPTS; ++i)
        m = fmaxf(m, x5[((size_t)b * NPTS + i) * 1024 + c]);
    gmax[t] = m;
}

__global__ void build_h16(const _Float16* __restrict__ feat16,  // [BN,192]
                          const float* __restrict__ gmax,       // [B,1024]
                          _Float16* __restrict__ h16) {         // [BN,1216]
    int t = blockIdx.x * blockDim.x + threadIdx.x;
    long total = (long)BN * 1216;
    if (t >= total) return;
    int row = t / 1216;
    int k = t - row * 1216;
    int b = row >> 11;   // / NPTS
    h16[t] = (k < 192) ? feat16[(size_t)row * 192 + k]
                       : (_Float16)gmax[(size_t)b * 1024 + (k - 192)];
}

__global__ void scatter_kernel(const float* __restrict__ h4,  // [BN,4]
                               float* __restrict__ out) {
    int t = blockIdx.x * blockDim.x + threadIdx.x;
    if (t >= BN) return;
#pragma unroll
    for (int c = 0; c < 4; ++c)
        out[(size_t)c * BN + t] = h4[(size_t)t * 4 + c];
}

// =====================================================================
// Host launch
// =====================================================================
extern "C" void kernel_launch(void* const* d_in, const int* in_sizes, int n_in,
                              void* d_out, int out_size, void* d_ws, size_t ws_size,
                              hipStream_t stream) {
    (void)in_sizes; (void)n_in; (void)out_size; (void)ws_size;

    const float* d   = (const float*)d_in[0];
    const float* x0  = (const float*)d_in[1];
    const float* r   = (const float*)d_in[2];
    const float* w1a = (const float*)d_in[3];  const float* p1a = (const float*)d_in[4];
    const float* w1b = (const float*)d_in[5];  const float* p1b = (const float*)d_in[6];
    const float* w2a = (const float*)d_in[7];  const float* p2a = (const float*)d_in[8];
    const float* w2b = (const float*)d_in[9];  const float* p2b = (const float*)d_in[10];
    const float* w3  = (const float*)d_in[11]; const float* p3  = (const float*)d_in[12];
    const float* w4  = (const float*)d_in[13]; const float* p4  = (const float*)d_in[14];
    const float* wl1 = (const float*)d_in[15]; const float* pl1 = (const float*)d_in[16];
    const float* wl2 = (const float*)d_in[17]; const float* pl2 = (const float*)d_in[18];
    const float* wl3 = (const float*)d_in[19]; const float* pl3 = (const float*)d_in[20];
    const float* wl4 = (const float*)d_in[21]; const float* pl4 = (const float*)d_in[22];

    float* out = (float*)d_out;
    float* ws  = (float*)d_ws;

    // ---- workspace carve-out, all offsets 256B aligned; total ~121 MB ----
    auto a64 = [](size_t x) { return (x + 63) & ~(size_t)63; };   // in float units
    size_t off = 0;
    float* li     = ws + off;          off = a64(off + (size_t)BN * 10);
    int*   idx    = (int*)(ws + off);  off = a64(off + (size_t)BN * KNN);
    float* feat   = ws + off;          off = a64(off + (size_t)BN * 192);
    _Float16* feat16 = (_Float16*)(ws + off); off = a64(off + (size_t)BN * 96);   // 192 h
    float* gmax   = ws + off;          off = a64(off + (size_t)BATCH * 1024);
    float* h4     = ws + off;          off = a64(off + (size_t)BN * 4);

    // f16 weights (padded): sizes in halves
    _Float16* w1a16 = (_Float16*)(ws + off); off = a64(off + (64 * 32) / 2);
    _Float16* w1b16 = (_Float16*)(ws + off); off = a64(off + (64 * 64) / 2);
    _Float16* w2a16 = (_Float16*)(ws + off); off = a64(off + (64 * 128) / 2);
    _Float16* w2b16 = (_Float16*)(ws + off); off = a64(off + (64 * 64) / 2);
    _Float16* w3_16 = (_Float16*)(ws + off); off = a64(off + (64 * 128) / 2);
    _Float16* w4_16 = (_Float16*)(ws + off); off = a64(off + (1024 * 192) / 2);
    _Float16* wl1_16 = (_Float16*)(ws + off); off = a64(off + (256 * 1216) / 2);
    _Float16* wl2_16 = (_Float16*)(ws + off); off = a64(off + (256 * 256) / 2);
    _Float16* wl3_16 = (_Float16*)(ws + off); off = a64(off + (128 * 256) / 2);
    _Float16* wl4_16 = (_Float16*)(ws + off); off = a64(off + (16 * 128) / 2);

    float* arena = ws + off;   // 26.3M floats, phase-reused
    // conv phase:
    _Float16* ef16   = (_Float16*)arena;                                 // <=ROWSE*128 h
    _Float16* eo1_16 = (_Float16*)(arena + (size_t)ROWSE * 64);          // ROWSE*64 h
    float*    eo2F   = arena + (size_t)ROWSE * 96;                       // ROWSE*64 f32
    // dense phase (conv buffers dead):
    float*    x5   = arena;                                              // BN*1024 f32
    _Float16* h16  = (_Float16*)arena;                                   // BN*1216 h
    _Float16* h1_16 = (_Float16*)(arena + (size_t)BN * 640);             // BN*256 h
    _Float16* h2_16 = (_Float16*)(arena + (size_t)BN * 768);             // BN*256 h
    _Float16* h3_16 = (_Float16*)(arena + (size_t)BN * 896);             // BN*128 h

    const int WPB = 4;   // waves per block (blockDim = 128)
    auto blocksFor = [&](int rows, int Npad, int NT) {
        int tiles = (rows / 16) * (Npad / (16 * NT));
        return (tiles + WPB - 1) / WPB;
    };
    auto ceil256 = [](long n) { return (int)((n + 255) / 256); };

    // ---- stage 0: lifted inputs + direct outputs; weight conversion ----
    init_kernel<<<ceil256(BN), 256, 0, stream>>>(d, x0, r, li, out);
    wconv_kernel<<<ceil256(64 * 32), 256, 0, stream>>>(w1a, w1a16, 64, 20, 64, 32);
    wconv_kernel<<<ceil256(64 * 64), 256, 0, stream>>>(w1b, w1b16, 64, 64, 64, 64);
    wconv_kernel<<<ceil256(64 * 128), 256, 0, stream>>>(w2a, w2a16, 64, 128, 64, 128);
    wconv_kernel<<<ceil256(64 * 64), 256, 0, stream>>>(w2b, w2b16, 64, 64, 64, 64);
    wconv_kernel<<<ceil256(64 * 128), 256, 0, stream>>>(w3, w3_16, 64, 128, 64, 128);
    wconv_kernel<<<ceil256(1024 * 192), 256, 0, stream>>>(w4, w4_16, 1024, 192, 1024, 192);
    wconv_kernel<<<ceil256(256 * 1216), 256, 0, stream>>>(wl1, wl1_16, 256, 1216, 256, 1216);
    wconv_kernel<<<ceil256(256 * 256), 256, 0, stream>>>(wl2, wl2_16, 256, 256, 256, 256);
    wconv_kernel<<<ceil256(128 * 256), 256, 0, stream>>>(wl3, wl3_16, 128, 256, 128, 256);
    wconv_kernel<<<ceil256(4 * 128), 256, 0, stream>>>(wl4, wl4_16, 4, 128, 16, 128);

    dim3 kg(NPTS / 128, BATCH);

    // ---- EdgeConv 1 (C=10, K padded to 32) ----
    knn_kernel<10><<<kg, 128, 0, stream>>>(li, 10, 0, idx);
    edge_gather<<<ceil256((long)ROWSE * 32), 256, 0, stream>>>(li, 10, 0, 10, 32, idx, ef16);
    gemm16<4><<<blocksFor(ROWSE, 64, 4), 128, 0, stream>>>(
        ef16, 32, w1a16, 32, 64, p1a, nullptr, 0, 64, eo1_16, 64, ROWSE);
    gemm16<4><<<blocksFor(ROWSE, 64, 4), 128, 0, stream>>>(
        eo1_16, 64, w1b16, 64, 64, p1b, eo2F, 64, 64, nullptr, 0, ROWSE);
    maxk_kernel<<<ceil256((long)BN * 64), 256, 0, stream>>>(eo2F, feat, feat16, 0);

    // ---- EdgeConv 2 (C=64) ----
    knn_kernel<64><<<kg, 128, 0, stream>>>(feat, 192, 0, idx);
    edge_gather<<<ceil256((long)ROWSE * 128), 256, 0, stream>>>(feat, 192, 0, 64, 128, idx, ef16);
    gemm16<4><<<blocksFor(ROWSE, 64, 4), 128, 0, stream>>>(
        ef16, 128, w2a16, 128, 64, p2a, nullptr, 0, 64, eo1_16, 64, ROWSE);
    gemm16<4><<<blocksFor(ROWSE, 64, 4), 128, 0, stream>>>(
        eo1_16, 64, w2b16, 64, 64, p2b, eo2F, 64, 64, nullptr, 0, ROWSE);
    maxk_kernel<<<ceil256((long)BN * 64), 256, 0, stream>>>(eo2F, feat, feat16, 64);

    // ---- EdgeConv 3 (C=64, single layer) ----
    knn_kernel<64><<<kg, 128, 0, stream>>>(feat, 192, 64, idx);
    edge_gather<<<ceil256((long)ROWSE * 128), 256, 0, stream>>>(feat, 192, 64, 64, 128, idx, ef16);
    gemm16<4><<<blocksFor(ROWSE, 64, 4), 128, 0, stream>>>(
        ef16, 128, w3_16, 128, 64, p3, eo2F, 64, 64, nullptr, 0, ROWSE);
    maxk_kernel<<<ceil256((long)BN * 64), 256, 0, stream>>>(eo2F, feat, feat16, 128);

    // ---- x5 = PReLU(feat @ w4^T) [BN,1024]; global max over N ----
    gemm16<4><<<blocksFor(BN, 1024, 4), 128, 0, stream>>>(
        feat16, 192, w4_16, 192, 1024, p4, x5, 1024, 1024, nullptr, 0, BN);
    gmax_kernel<<<ceil256(BATCH * 1024), 256, 0, stream>>>(x5, gmax);

    // ---- MLP head (h materialized f16; arena reused after x5 dies) ----
    build_h16<<<ceil256((long)BN * 1216), 256, 0, stream>>>(feat16, gmax, h16);
    gemm16<4><<<blocksFor(BN, 256, 4), 128, 0, stream>>>(
        h16, 1216, wl1_16, 1216, 256, pl1, nullptr, 0, 256, h1_16, 256, BN);
    gemm16<4><<<blocksFor(BN, 256, 4), 128, 0, stream>>>(
        h1_16, 256, wl2_16, 256, 256, pl2, nullptr, 0, 256, h2_16, 256, BN);
    gemm16<4><<<blocksFor(BN, 128, 4), 128, 0, stream>>>(
        h2_16, 256, wl3_16, 256, 128, pl3, nullptr, 0, 128, h3_16, 128, BN);
    gemm16<1><<<blocksFor(BN, 16, 1), 128, 0, stream>>>(
        h3_16, 128, wl4_16, 128, 16, pl4, h4, 4, 4, nullptr, 0, BN);

    scatter_kernel<<<ceil256(BN), 256, 0, stream>>>(h4, out);
}
